// conduit_node_learning_9852654977699
// MI455X (gfx1250) — compile-verified, matrix operand order checked by reading, not run on hardware
//
#include <hip/hip_runtime.h>

#define PRE  128
#define NEXT 128

typedef __attribute__((ext_vector_type(16))) __bf16 v16bf;
typedef __attribute__((ext_vector_type(8)))  float  v8f;

union FragBF16 {
  v16bf v;
  unsigned int u[8];
  uint4 q[2];
};

// RNE f32 -> bf16 (used only in the bandwidth-bound prep kernels)
__device__ __forceinline__ unsigned int bgl_pack2_rne(float a, float b) {
  unsigned int ua = __float_as_uint(a), ub = __float_as_uint(b);
  ua = (ua + 0x7FFFu + ((ua >> 16) & 1u)) >> 16;
  ub = (ub + 0x7FFFu + ((ub >> 16) & 1u)) >> 16;
  return (ua & 0xFFFFu) | (ub << 16);
}

__device__ __forceinline__ unsigned short bgl_pack1_rne(float a) {
  unsigned int ua = __float_as_uint(a);
  return (unsigned short)((ua + 0x7FFFu + ((ua >> 16) & 1u)) >> 16);
}

// Single v_perm_b32: D = { hi16(hi), hi16(lo) }  (truncating f32->bf16, fallback path)
__device__ __forceinline__ unsigned int bgl_trunc2(float lo, float hi) {
  return __builtin_amdgcn_perm(__float_as_uint(hi), __float_as_uint(lo), 0x07060302u);
}

// CDNA5 packed bf16 add (VOP3P op 35)
__device__ __forceinline__ unsigned int bgl_pk_add_bf16(unsigned int a, unsigned int b) {
  unsigned int d;
  asm("v_pk_add_bf16 %0, %1, %2" : "=v"(d) : "v"(a), "v"(b));
  return d;
}

__device__ __forceinline__ float4 f4add(float4 a, float4 b) {
  return make_float4(a.x + b.x, a.y + b.y, a.z + b.z, a.w + b.w);
}

// --- Prep 1: three 128x128 f32 weight matrices -> bf16 (RNE) ---
__global__ __launch_bounds__(256) void bgl_convert_w_kernel(
    const float* __restrict__ wl, const float* __restrict__ wr,
    const float* __restrict__ wu, unsigned short* __restrict__ out) {
  int i = blockIdx.x * blockDim.x + threadIdx.x;
  if (i < PRE * NEXT) {
    out[i]                  = bgl_pack1_rne(wl[i]);
    out[PRE * NEXT + i]     = bgl_pack1_rne(wr[i]);
    out[2 * PRE * NEXT + i] = bgl_pack1_rne(wu[i]);
  }
}

// --- Prep 2: node embedding table f32 -> bf16 (RNE), 2 elements/thread ---
__global__ __launch_bounds__(256) void bgl_convert_emb_kernel(
    const float* __restrict__ emb, unsigned int* __restrict__ out, int n_pairs) {
  int i = blockIdx.x * blockDim.x + threadIdx.x;
  if (i < n_pairs) {
    float2 f = ((const float2*)emb)[i];
    out[i] = bgl_pack2_rne(f.x, f.y);
  }
}

// --- Main: one wave32 handles 16 edges x 128 outputs via v_wmma_f32_16x16x32_bf16 ---
template <bool PREBF>
__global__ __launch_bounds__(256) void bgl_edge_gate_kernel(
    const float* __restrict__ embf32,
    const unsigned short* __restrict__ embbf,
    const long long* __restrict__ edges,
    const unsigned short* __restrict__ wbf,   // [3][NEXT][PRE] bf16
    const float* __restrict__ bupd,
    float* __restrict__ out,
    int n_edges, int n_tiles) {
  const int lane = threadIdx.x & 31;
  const int wave = threadIdx.x >> 5;
  const int tile = blockIdx.x * (blockDim.x >> 5) + wave;
  if (tile >= n_tiles) return;   // wave-uniform: EXEC stays all-ones for WMMA

  const int m    = lane & 15;    // edge row within tile / column within N-tile
  const int half = lane >> 4;    // K-half selector per ISA 16-bit A/B layout
  const long long ebase = (long long)tile * 16;

  long long eidx = ebase + m;
  if (eidx >= n_edges) eidx = n_edges - 1;   // clamp gather (tail safety)
  const longlong2 epair = *(const longlong2*)(edges + 2 * eidx);
  const long long src = epair.x, dst = epair.y;

  // A fragments for 4 K-tiles: x0, x1, x0+x1.
  // Lane holds K in [kt*32 + half*8, +8) and [kt*32 + 16 + half*8, +8).
  FragBF16 a0[4], a1[4], as_[4];
  if (PREBF) {
    const unsigned short* r0 = embbf + src * PRE;
    const unsigned short* r1 = embbf + dst * PRE;
#pragma unroll
    for (int kt = 0; kt < 4; ++kt) {
      const int base = kt * 32 + half * 8;
      a0[kt].q[0] = *(const uint4*)(r0 + base);
      a0[kt].q[1] = *(const uint4*)(r0 + base + 16);
      a1[kt].q[0] = *(const uint4*)(r1 + base);
      a1[kt].q[1] = *(const uint4*)(r1 + base + 16);
#pragma unroll
      for (int j = 0; j < 8; ++j)
        as_[kt].u[j] = bgl_pk_add_bf16(a0[kt].u[j], a1[kt].u[j]);
    }
  } else {
    const float* r0 = embf32 + src * PRE;
    const float* r1 = embf32 + dst * PRE;
#pragma unroll
    for (int kt = 0; kt < 4; ++kt) {
      const int base = kt * 32 + half * 8;
      const float4* q0 = (const float4*)(r0 + base);
      const float4* q1 = (const float4*)(r1 + base);
      float4 c0[4] = {q0[0], q0[1], q0[4], q0[5]};
      float4 c1[4] = {q1[0], q1[1], q1[4], q1[5]};
#pragma unroll
      for (int j = 0; j < 4; ++j) {
        float4 s = f4add(c0[j], c1[j]);
        a0[kt].u[2 * j]     = bgl_trunc2(c0[j].x, c0[j].y);
        a0[kt].u[2 * j + 1] = bgl_trunc2(c0[j].z, c0[j].w);
        a1[kt].u[2 * j]     = bgl_trunc2(c1[j].x, c1[j].y);
        a1[kt].u[2 * j + 1] = bgl_trunc2(c1[j].z, c1[j].w);
        as_[kt].u[2 * j]     = bgl_trunc2(s.x, s.y);
        as_[kt].u[2 * j + 1] = bgl_trunc2(s.z, s.w);
      }
    }
  }

#pragma unroll
  for (int nt = 0; nt < 8; ++nt) {
    v8f accL = {}, accR = {}, accU = {};
    // B fragment: lane holds column n = nt*16 + m of B, i.e. row n of W,
    // same K pattern as A (ISA 16-bit B layout mirrors A with lane<->N).
    const int nrow = nt * 16 + m;
#pragma unroll
    for (int kt = 0; kt < 4; ++kt) {
      const int boff = nrow * PRE + kt * 32 + half * 8;
      FragBF16 bl, br, bu;
      bl.q[0] = *(const uint4*)(wbf + boff);
      bl.q[1] = *(const uint4*)(wbf + boff + 16);
      br.q[0] = *(const uint4*)(wbf + PRE * NEXT + boff);
      br.q[1] = *(const uint4*)(wbf + PRE * NEXT + boff + 16);
      bu.q[0] = *(const uint4*)(wbf + 2 * PRE * NEXT + boff);
      bu.q[1] = *(const uint4*)(wbf + 2 * PRE * NEXT + boff + 16);

      accL = __builtin_amdgcn_wmma_f32_16x16x32_bf16(
          false, a0[kt].v, false, bl.v, (short)0, accL, false, false);
      accR = __builtin_amdgcn_wmma_f32_16x16x32_bf16(
          false, a1[kt].v, false, br.v, (short)0, accR, false, false);
      accU = __builtin_amdgcn_wmma_f32_16x16x32_bf16(
          false, as_[kt].v, false, bu.v, (short)0, accU, false, false);
    }

    // Epilogue. C/D layout: VGPR r holds rows (r, r+8), lane&15 is column.
    // (sig(a)+sig(b))*tanh(c) = (2+ea+eb)(1-ec) / [(1+ea)(1+eb)(1+ec)]
    // with ea=e^-a, eb=e^-b, ec=e^-2c  ->  3x v_exp + 1x v_rcp per output.
    const float bv = bupd[nrow];
#pragma unroll
    for (int r = 0; r < 8; ++r) {
      const long long erow = ebase + r + half * 8;
      if (erow < n_edges) {
        float ea = __expf(-accL[r]);
        float eb = __expf(-accR[r]);
        float ec = __expf(-2.f * (accU[r] + bv));
        float num = (2.f + ea + eb) * (1.f - ec);
        float den = (1.f + ea) * ((1.f + eb) * (1.f + ec));
        out[erow * NEXT + nrow] = num * __builtin_amdgcn_rcpf(den);
      }
    }
  }
}

extern "C" void kernel_launch(void* const* d_in, const int* in_sizes, int n_in,
                              void* d_out, int out_size, void* d_ws, size_t ws_size,
                              hipStream_t stream) {
  const float*      emb   = (const float*)d_in[0];
  const long long*  edges = (const long long*)d_in[1];
  const float*      wl    = (const float*)d_in[2];
  const float*      wr    = (const float*)d_in[3];
  const float*      wu    = (const float*)d_in[4];
  const float*      bu    = (const float*)d_in[5];
  float*            out   = (float*)d_out;

  unsigned short* wbf = (unsigned short*)d_ws;            // 96 KB
  unsigned short* ebf = wbf + 3 * PRE * NEXT;             // bf16 embedding table

  const long long n_emb   = in_sizes[0];                  // n_nodes * 128
  const int       n_edges = in_sizes[1] / 2;
  const int       n_tiles = (n_edges + 15) / 16;
  const size_t    needed  = (size_t)3 * PRE * NEXT * 2 + (size_t)n_emb * 2;
  const bool      prebf   = ws_size >= needed;

  bgl_convert_w_kernel<<<dim3((PRE * NEXT + 255) / 256), dim3(256), 0, stream>>>(
      wl, wr, wu, wbf);

  if (prebf) {
    const int n_pairs = (int)(n_emb / 2);
    bgl_convert_emb_kernel<<<dim3((n_pairs + 255) / 256), dim3(256), 0, stream>>>(
        emb, (unsigned int*)ebf, n_pairs);
  }

  const int waves_per_block = 8;   // 256 threads = 8 wave32
  dim3 grid((n_tiles + waves_per_block - 1) / waves_per_block);
  if (prebf) {
    bgl_edge_gate_kernel<true><<<grid, dim3(256), 0, stream>>>(
        emb, ebf, edges, wbf, bu, out, n_edges, n_tiles);
  } else {
    bgl_edge_gate_kernel<false><<<grid, dim3(256), 0, stream>>>(
        emb, ebf, edges, wbf, bu, out, n_edges, n_tiles);
  }
}
